// MaskedCrossAttention_74947179315323
// MI455X (gfx1250) — compile-verified, matrix operand order checked by reading, not run on hardware
//
#include <hip/hip_runtime.h>
#include <hip/hip_bf16.h>
#include <math.h>

// ---------------------------------------------------------------------------
// MaskedCrossAttention forward for MI455X (gfx1250), wave32 + WMMA f16.
//   B=8, T=1024, DIM=2048, S=T_img*n=512, H=16, D=64, HIDDEN=1024
// GEMMs: v_wmma_f32_16x16x32_f16 + double-buffered LDS staged with
// global_load_async_to_lds_b128 (ASYNCcnt path). Attention is fused
// flash-style with the transposed-operand trick (lane-local softmax).
// ---------------------------------------------------------------------------

typedef _Float16 half16 __attribute__((ext_vector_type(16)));
typedef _Float16 half8  __attribute__((ext_vector_type(8)));
typedef float    float8 __attribute__((ext_vector_type(8)));

#define B_   8
#define T_   1024
#define DIM_ 2048
#define S_   512
#define H_   16
#define D_   64
#define HID_ 1024

static __device__ inline float8 wmma16(half16 a, half16 b, float8 c) {
  // D = A(16x32 f16) * B(32x16 f16) + C(16x16 f32)
  return __builtin_amdgcn_wmma_f32_16x16x32_f16(
      /*neg_a=*/false, a, /*neg_b=*/false, b,
      /*c_mod=*/(short)0, c, /*reuse_a=*/false, /*reuse_b=*/false);
}

// Build a 16-half fragment from two contiguous 16-byte groups (lo: halves
// h=0..7, hi: halves h=8..15), matching the CDNA5 A/B K-packing
// K(lane,h) = 8*(lane>>4) + (h<8 ? h : h+8).
static __device__ inline half16 make_frag(const _Float16* lo, const _Float16* hi) {
  half8 l = *(const half8*)lo;
  half8 h = *(const half8*)hi;
  half16 f;
#pragma unroll
  for (int i = 0; i < 8; ++i) { f[i] = l[i]; f[i + 8] = h[i]; }
  return f;
}

// CDNA5 async DMA: 16 bytes global -> LDS, tracked by ASYNCcnt.
// lds_off = LDS byte address (low 32 bits of the generic shared pointer).
static __device__ inline void async_ld16(unsigned lds_off, const _Float16* src) {
  asm volatile("global_load_async_to_lds_b128 %0, %1, off"
               :: "v"(lds_off), "v"(src)
               : "memory");
}
static __device__ inline void wait_async0() {
  asm volatile("s_wait_asynccnt 0x0" ::: "memory");
}
static __device__ inline unsigned lds_off_of(const void* p) {
  return (unsigned)(uintptr_t)p;  // LDS aperture: offset lives in addr[31:0]
}

// ---------------------------------------------------------------------------
// LayerNorm over DIM=2048 + cast to f16. One block (256 thr) per row.
// ---------------------------------------------------------------------------
__global__ __launch_bounds__(256) void ln_kernel(const float* __restrict__ x,
                                                 const float* __restrict__ gamma,
                                                 const float* __restrict__ beta,
                                                 _Float16* __restrict__ xn) {
  const int row = blockIdx.x;
  const int tid = threadIdx.x;
  const float* xr = x + (size_t)row * DIM_;
  float v[8], s = 0.f, ss = 0.f;
#pragma unroll
  for (int j = 0; j < 8; ++j) {
    v[j] = xr[tid + j * 256];
    s += v[j];
    ss += v[j] * v[j];
  }
#pragma unroll
  for (int o = 16; o >= 1; o >>= 1) {
    s += __shfl_xor(s, o, 32);
    ss += __shfl_xor(ss, o, 32);
  }
  __shared__ float red[16];
  if ((tid & 31) == 0) { red[tid >> 5] = s; red[8 + (tid >> 5)] = ss; }
  __syncthreads();
  float S = 0.f, SS = 0.f;
#pragma unroll
  for (int w = 0; w < 8; ++w) { S += red[w]; SS += red[8 + w]; }
  const float mu = S * (1.f / DIM_);
  const float var = SS * (1.f / DIM_) - mu * mu;
  const float rs = rsqrtf(var + 1e-5f);
#pragma unroll
  for (int j = 0; j < 8; ++j) {
    const int c = tid + j * 256;
    xn[(size_t)row * DIM_ + c] = (_Float16)((v[j] - mu) * rs * gamma[c] + beta[c]);
  }
}

// ---------------------------------------------------------------------------
// Transpose + cast f32 W[K][N] -> f16 WT[N][K] (32x32 LDS tiles).
// ---------------------------------------------------------------------------
__global__ void tcast_kernel(const float* __restrict__ W, _Float16* __restrict__ WT,
                             int K, int N) {
  __shared__ float t[32][33];
  const int tx = threadIdx.x, ty = threadIdx.y;
  const int n0 = blockIdx.x * 32, k0 = blockIdx.y * 32;
#pragma unroll
  for (int j = 0; j < 32; j += 8)
    t[ty + j][tx] = W[(size_t)(k0 + ty + j) * N + (n0 + tx)];
  __syncthreads();
#pragma unroll
  for (int j = 0; j < 32; j += 8)
    WT[(size_t)(n0 + ty + j) * K + (k0 + tx)] = (_Float16)t[tx][ty + j];
}

// Elementwise f32 -> f16 cast.
__global__ void cast16_kernel(const float* __restrict__ in, _Float16* __restrict__ out,
                              int n) {
  const int i = (blockIdx.x * 256 + threadIdx.x) * 4;
#pragma unroll
  for (int j = 0; j < 4; ++j)
    if (i + j < n) out[i + j] = (_Float16)in[i + j];
}

// ---------------------------------------------------------------------------
// WMMA GEMM: C[M,N] = A[M,K] (row-major f16) * Bc[N,K] (column-major f16).
// Block tile 128x128, 8 waves (2x4), k-step 32. LDS is double-buffered and
// filled by global_load_async_to_lds_b128; one s_wait_asynccnt + one barrier
// per k-step. LDS rows padded to 40 halves (80 B) so the ds_read_b128 frag
// reads are bank-conflict-free.
// Epilogues: 0 = f16 row-major, 1 = KV scatter (K row-major + V transposed),
//            2 = f32 row-major.
// ---------------------------------------------------------------------------
template <int EPI>
__global__ __launch_bounds__(256) void gemm_wmma_k(
    const _Float16* __restrict__ A, const _Float16* __restrict__ Bc,
    int M, int N, int K,
    float* __restrict__ Cf, _Float16* __restrict__ Ch,
    _Float16* __restrict__ kbuf, _Float16* __restrict__ vt) {
  __shared__ __align__(16) _Float16 sA[2][128 * 40];
  __shared__ __align__(16) _Float16 sB[2][128 * 40];
  const int tid = threadIdx.x;
  const int lane = tid & 31, wave = tid >> 5;
  const int wm = wave >> 2, wn = wave & 3;     // 2 x 4 wave grid
  const int l16 = lane & 15, hh = lane >> 4;
  const int mBlk = blockIdx.y * 128, nBlk = blockIdx.x * 128;

  // cooperative tile-load mapping: thread -> (row, 16-half column group)
  const int lr = tid >> 1;
  const int lc = (tid & 1) * 16;
  const _Float16* gA = A + (size_t)(mBlk + lr) * K + lc;
  const _Float16* gB = Bc + (size_t)(nBlk + lr) * K + lc;
  const unsigned oA[2] = {lds_off_of(&sA[0][lr * 40 + lc]),
                          lds_off_of(&sA[1][lr * 40 + lc])};
  const unsigned oB[2] = {lds_off_of(&sB[0][lr * 40 + lc]),
                          lds_off_of(&sB[1][lr * 40 + lc])};

  float8 acc[4][2];
#pragma unroll
  for (int mt = 0; mt < 4; ++mt)
#pragma unroll
    for (int nt = 0; nt < 2; ++nt)
#pragma unroll
      for (int i = 0; i < 8; ++i) acc[mt][nt][i] = 0.f;

  // prologue: stage k-step 0 into buffer 0
  async_ld16(oA[0], gA);      async_ld16(oA[0] + 16, gA + 8);
  async_ld16(oB[0], gB);      async_ld16(oB[0] + 16, gB + 8);

  int buf = 0;
  for (int k0 = 0; k0 < K; k0 += 32) {
    wait_async0();            // this wave's DMA into buf landed
    __syncthreads();          // everyone's DMA into buf landed
    if (k0 + 32 < K) {        // prefetch next k-step into the other buffer
      const int nb = buf ^ 1;
      async_ld16(oA[nb], gA + k0 + 32);      async_ld16(oA[nb] + 16, gA + k0 + 40);
      async_ld16(oB[nb], gB + k0 + 32);      async_ld16(oB[nb] + 16, gB + k0 + 40);
    }
    const _Float16* bufA = sA[buf];
    const _Float16* bufB = sB[buf];

    half16 af[4], bf[2];
#pragma unroll
    for (int mt = 0; mt < 4; ++mt) {
      const _Float16* p = bufA + (wm * 64 + mt * 16 + l16) * 40 + hh * 8;
      af[mt] = make_frag(p, p + 16);
    }
#pragma unroll
    for (int nt = 0; nt < 2; ++nt) {
      const _Float16* p = bufB + (wn * 32 + nt * 16 + l16) * 40 + hh * 8;
      bf[nt] = make_frag(p, p + 16);
    }
#pragma unroll
    for (int mt = 0; mt < 4; ++mt)
#pragma unroll
      for (int nt = 0; nt < 2; ++nt)
        acc[mt][nt] = wmma16(af[mt], bf[nt], acc[mt][nt]);
    buf ^= 1;
  }

#pragma unroll
  for (int mt = 0; mt < 4; ++mt)
#pragma unroll
    for (int nt = 0; nt < 2; ++nt)
#pragma unroll
      for (int i = 0; i < 8; ++i) {
        const int m = mBlk + wm * 64 + mt * 16 + hh * 8 + i;
        const int n = nBlk + wn * 32 + nt * 16 + l16;
        const float v = acc[mt][nt][i];
        if (EPI == 0) {
          Ch[(size_t)m * N + n] = (_Float16)v;
        } else if (EPI == 2) {
          Cf[(size_t)m * N + n] = v;
        } else {  // KV scatter: m = b*512+s ; n<1024 -> K, else -> V^T
          const int b = m >> 9, s = m & 511;
          if (n < HID_) {
            const int hI = n >> 6, d = n & 63;
            kbuf[((size_t)(b * H_ + hI) * S_ + s) * D_ + d] = (_Float16)v;
          } else {
            const int n2 = n - HID_;
            const int hI = n2 >> 6, d = n2 & 63;
            vt[((size_t)(b * H_ + hI) * D_ + d) * S_ + s] = (_Float16)v;
          }
        }
      }
}

// ---------------------------------------------------------------------------
// Fused flash attention, transposed-operand form (all lane-local softmax):
//   S^T tile = K_chunk(16x64) * Q^T(64x16)  -> lane owns one t-column
//   O^T tile = V^T_chunk(16x32) * P^T(32x16)
// 1 wave = 16 query rows; block = 8 waves = 128 rows; grid = B*H*(T/128).
// Reference quirk: softmax of UNSCALED scores, probs divided by SCALE=8.
// ---------------------------------------------------------------------------
__global__ __launch_bounds__(256) void fattn_kernel(const _Float16* __restrict__ Q,
                                                    const _Float16* __restrict__ Kb,
                                                    const _Float16* __restrict__ Vt,
                                                    _Float16* __restrict__ O) {
  const int lane = threadIdx.x & 31, wave = threadIdx.x >> 5;
  const int l16 = lane & 15, hh = lane >> 4;
  const int bh = blockIdx.x >> 3;
  const int b = bh >> 4, h = bh & 15;
  const int t = (blockIdx.x & 7) * 128 + wave * 16 + l16;  // this lane's t

  const _Float16* qrow = Q + ((size_t)b * T_ + t) * HID_ + h * D_;
  half16 qf[2];
  qf[0] = make_frag(qrow + hh * 8, qrow + hh * 8 + 16);            // d 0..31
  qf[1] = make_frag(qrow + 32 + hh * 8, qrow + 32 + hh * 8 + 16);  // d 32..63

  const _Float16* Kp = Kb + (size_t)(b * H_ + h) * S_ * D_;    // [s][d]
  const _Float16* Vp = Vt + (size_t)(b * H_ + h) * D_ * S_;    // [d][s]

  float mrun = -1e30f, lrun = 0.f;
  float8 oacc[4];
#pragma unroll
  for (int dt = 0; dt < 4; ++dt)
#pragma unroll
    for (int i = 0; i < 8; ++i) oacc[dt][i] = 0.f;

  for (int sc = 0; sc < S_; sc += 32) {
    float8 st0, st1;
#pragma unroll
    for (int i = 0; i < 8; ++i) { st0[i] = 0.f; st1[i] = 0.f; }
    {
      const _Float16* kp0 = Kp + (size_t)(sc + l16) * D_ + hh * 8;
      const _Float16* kp1 = Kp + (size_t)(sc + 16 + l16) * D_ + hh * 8;
      st0 = wmma16(make_frag(kp0, kp0 + 16), qf[0], st0);
      st0 = wmma16(make_frag(kp0 + 32, kp0 + 48), qf[1], st0);
      st1 = wmma16(make_frag(kp1, kp1 + 16), qf[0], st1);
      st1 = wmma16(make_frag(kp1 + 32, kp1 + 48), qf[1], st1);
    }
    // st0[i] = S[t][sc + 8*hh + i], st1[i] = S[t][sc+16 + 8*hh + i]
    float cm = -1e30f;
#pragma unroll
    for (int i = 0; i < 8; ++i) cm = fmaxf(cm, fmaxf(st0[i], st1[i]));
    cm = fmaxf(cm, __shfl_xor(cm, 16, 32));
    const float mnew = fmaxf(mrun, cm);
    const float fac = __expf(mrun - mnew);

    half16 pf;            // P^T B-fragment: pf[h] <-> s = kmap(lane,h)
    float cs = 0.f;
#pragma unroll
    for (int i = 0; i < 8; ++i) {
      const float e = __expf(st0[i] - mnew);
      cs += e;
      pf[i] = (_Float16)e;
    }
#pragma unroll
    for (int i = 0; i < 8; ++i) {
      const float e = __expf(st1[i] - mnew);
      cs += e;
      pf[8 + i] = (_Float16)e;
    }
    cs += __shfl_xor(cs, 16, 32);
    lrun = lrun * fac + cs;
    mrun = mnew;

#pragma unroll
    for (int dt = 0; dt < 4; ++dt) {
#pragma unroll
      for (int i = 0; i < 8; ++i) oacc[dt][i] *= fac;
      const _Float16* vp = Vp + (size_t)(dt * 16 + l16) * S_ + sc + hh * 8;
      oacc[dt] = wmma16(make_frag(vp, vp + 16), pf, oacc[dt]);
    }
  }

  const float rl = 1.0f / (lrun * 8.0f);  // softmax norm * (1/SCALE)
  _Float16* orow = O + ((size_t)b * T_ + t) * HID_ + h * D_;
#pragma unroll
  for (int dt = 0; dt < 4; ++dt) {
    half8 ov;
#pragma unroll
    for (int i = 0; i < 8; ++i) ov[i] = (_Float16)(oacc[dt][i] * rl);
    *(half8*)(orow + dt * 16 + hh * 8) = ov;  // d = dt*16 + 8*hh + i
  }
}

// ---------------------------------------------------------------------------
extern "C" void kernel_launch(void* const* d_in, const int* in_sizes, int n_in,
                              void* d_out, int out_size, void* d_ws, size_t ws_size,
                              hipStream_t stream) {
  const float* x     = (const float*)d_in[0];
  const float* media = (const float*)d_in[1];
  // d_in[2]: media_locations — mask is discarded by the reference; unused.
  const float* Wq    = (const float*)d_in[3];
  const float* Wkv   = (const float*)d_in[4];
  const float* Wo    = (const float*)d_in[5];
  const float* gamma = (const float*)d_in[6];
  const float* beta  = (const float*)d_in[7];
  float* out = (float*)d_out;

  char* ws = (char*)d_ws;
  _Float16* xn   = (_Float16*)ws; ws += (size_t)B_ * T_ * DIM_ * 2;        // 32 MB
  _Float16* WqT  = (_Float16*)ws; ws += (size_t)HID_ * DIM_ * 2;           // 4 MB
  _Float16* WkvT = (_Float16*)ws; ws += (size_t)DIM_ * HID_ * 2;           // 4 MB
  _Float16* WoT  = (_Float16*)ws; ws += (size_t)DIM_ * HID_ * 2;           // 4 MB
  _Float16* md16 = (_Float16*)ws; ws += (size_t)B_ * S_ * HID_ * 2;        // 8 MB
  _Float16* Qb   = (_Float16*)ws; ws += (size_t)B_ * T_ * HID_ * 2;        // 16 MB
  _Float16* Kbf  = (_Float16*)ws; ws += (size_t)B_ * H_ * S_ * D_ * 2;     // 8 MB
  _Float16* Vtb  = (_Float16*)ws; ws += (size_t)B_ * H_ * D_ * S_ * 2;     // 8 MB
  _Float16* Ob   = (_Float16*)ws; ws += (size_t)B_ * T_ * HID_ * 2;        // 16 MB

  // 1) LayerNorm + cast
  ln_kernel<<<B_ * T_, 256, 0, stream>>>(x, gamma, beta, xn);
  // 2) Weight transpose-casts -> column-major f16 B operands
  tcast_kernel<<<dim3(HID_ / 32, DIM_ / 32), dim3(32, 8), 0, stream>>>(Wq, WqT, DIM_, HID_);
  tcast_kernel<<<dim3(DIM_ / 32, HID_ / 32), dim3(32, 8), 0, stream>>>(Wkv, WkvT, HID_, DIM_);
  tcast_kernel<<<dim3(DIM_ / 32, HID_ / 32), dim3(32, 8), 0, stream>>>(Wo, WoT, HID_, DIM_);
  // 3) media cast
  const int nMedia = B_ * S_ * HID_;  // 4,194,304
  cast16_kernel<<<nMedia / 1024, 256, 0, stream>>>(media, md16, nMedia);
  // 4) Q = xn @ Wq            (M=8192, N=1024, K=2048) -> f16 [b*T+t][h*64+d]
  gemm_wmma_k<0><<<dim3(HID_ / 128, (B_ * T_) / 128), 256, 0, stream>>>(
      xn, WqT, B_ * T_, HID_, DIM_, nullptr, Qb, nullptr, nullptr);
  // 5) KV = media @ Wkv       (M=4096, N=2048, K=1024) -> K[b,h,s,d], V^T[b,h,d,s]
  gemm_wmma_k<1><<<dim3(DIM_ / 128, (B_ * S_) / 128), 256, 0, stream>>>(
      md16, WkvT, B_ * S_, DIM_, HID_, nullptr, nullptr, Kbf, Vtb);
  // 6) Fused flash attention -> O f16 [b*T+t][h*64+d]
  fattn_kernel<<<B_ * H_ * (T_ / 128), 256, 0, stream>>>(Qb, Kbf, Vtb, Ob);
  // 7) out = O @ Wo           (M=8192, N=2048, K=1024) -> f32 d_out
  gemm_wmma_k<2><<<dim3(DIM_ / 128, (B_ * T_) / 128), 256, 0, stream>>>(
      Ob, WoT, B_ * T_, DIM_, HID_, out, nullptr, nullptr, nullptr);
}
